// VectorQuantizer_120259084857
// MI455X (gfx1250) — matrix operand order repaired and checked by reading, not compiled, not run on hardware
//
#include <hip/hip_runtime.h>
#include <stdint.h>

typedef __attribute__((ext_vector_type(16))) __bf16 v16bf;
typedef __attribute__((ext_vector_type(8)))  float  v8f;

#define D_DIM   256
#define K_CODES 1024
#define N_ROWS  65536
#define N_FLAT  (N_ROWS * D_DIM)   // 16777216 output elements + 1 loss scalar
#define MWAVE   32                 // rows per wave (2 x 16-row strips)

union V16 { v16bf v; uint32_t u[8]; };

__device__ __forceinline__ uint32_t f32_to_bf16(float f) {
  uint32_t u = __float_as_uint(f);
  return (u + 0x7FFFu + ((u >> 16) & 1u)) >> 16;   // round-to-nearest-even
}
__device__ __forceinline__ float bf16_to_f32(uint32_t b) {
  return __uint_as_float(b << 16);
}

// ---------------------------------------------------------------------------
// Prep 1: planar bf16 hi/lo codebook, pre-swizzled into WMMA-B operand order.
// word index w = ((n*8 + c)*2 + g)*8 + v  holds pack(k, k+1), k = c*32+(v>>2)*16+g*8+(v&3)*2
// -> the 8 words of one B operand (fixed n,c,g) are 32B-contiguous, so the hot
//    loop loads each operand as one v16bf (2 x global_load_b128, in place).
// ---------------------------------------------------------------------------
__global__ void vq_pack_codebook(const float* __restrict__ cb,
                                 uint32_t* __restrict__ cbh,
                                 uint32_t* __restrict__ cbl) {
  int w = blockIdx.x * blockDim.x + threadIdx.x;   // 0 .. 131071
  int v = w & 7, g = (w >> 3) & 1, c = (w >> 4) & 7, n = w >> 7;
  int k = (c << 5) + ((v >> 2) << 4) + (g << 3) + ((v & 3) << 1);
  const float* s = cb + (size_t)n * D_DIM + k;
  float x0 = s[0], x1 = s[1];
  uint32_t h0 = f32_to_bf16(x0), h1 = f32_to_bf16(x1);
  uint32_t l0 = f32_to_bf16(x0 - bf16_to_f32(h0));
  uint32_t l1 = f32_to_bf16(x1 - bf16_to_f32(h1));
  cbh[w] = h0 | (h1 << 16);
  cbl[w] = l0 | (l1 << 16);
}

// ---------------------------------------------------------------------------
// Prep 2: bias[n] = -0.5*||e_n||^2 (exact fp32), and zero the loss accumulator.
// ---------------------------------------------------------------------------
__global__ void vq_bias(const float* __restrict__ cb, float* __restrict__ bias,
                        double* __restrict__ lacc) {
  int n = blockIdx.x * blockDim.x + threadIdx.x;   // 0 .. 1023
  const float4* r = (const float4*)(cb + (size_t)n * D_DIM);
  float s = 0.f;
  for (int d = 0; d < D_DIM / 4; ++d) {
    float4 v = r[d];
    s += v.x * v.x + v.y * v.y + v.z * v.z + v.w * v.w;
  }
  bias[n] = -0.5f * s;
  if (n == 0) *lacc = 0.0;
}

// ---------------------------------------------------------------------------
// Main: one wave per 32-row block (two 16-row WMMA strips sharing each B tile).
// score(m,k) = x.e_k - 0.5||e_k||^2 via 3-term bf16-split, 6 independent
// accumulator chains; running per-lane argmax; butterfly reduce; gather+write.
// ---------------------------------------------------------------------------
__launch_bounds__(32)
__global__ void vq_main(const float* __restrict__ lat, const float* __restrict__ cb,
                        const v16bf* __restrict__ cbh, const v16bf* __restrict__ cbl,
                        const float* __restrict__ bias,
                        float* __restrict__ out, double* __restrict__ lacc) {
  const int lane = threadIdx.x;
  const int rit  = lane & 15;
  const int grp  = lane >> 4;
  const int row0 = blockIdx.x * MWAVE;

  // ---- A tiles (2 strips x 16 rows x 256), bf16 hi/lo split, VGPR-resident --
  // A 16x32 bf16 layout: lane m holds k = (v>=4?16:0) + grp*8 + (v&3)*2 (+1)
  v16bf a_hi[2][8], a_lo[2][8];
#pragma unroll
  for (int s = 0; s < 2; ++s) {
    const float* arow = lat + (size_t)(row0 + (s << 4) + rit) * D_DIM;
#pragma unroll
    for (int c = 0; c < 8; ++c) {
      V16 th, tl;
#pragma unroll
      for (int v = 0; v < 8; ++v) {
        int k = (c << 5) + ((v >> 2) << 4) + (grp << 3) + ((v & 3) << 1);
        float x0 = arow[k], x1 = arow[k + 1];
        uint32_t h0 = f32_to_bf16(x0), h1 = f32_to_bf16(x1);
        uint32_t l0 = f32_to_bf16(x0 - bf16_to_f32(h0));
        uint32_t l1 = f32_to_bf16(x1 - bf16_to_f32(h1));
        th.u[v] = h0 | (h1 << 16);
        tl.u[v] = l0 | (l1 << 16);
      }
      a_hi[s][c] = th.v;
      a_lo[s][c] = tl.v;
    }
  }

  float best_val[2][8];
  int   best_idx[2][8];
#pragma unroll
  for (int s = 0; s < 2; ++s)
#pragma unroll
    for (int r = 0; r < 8; ++r) { best_val[s][r] = -3.4e38f; best_idx[s][r] = 0; }

  for (int t = 0; t < (K_CODES / 16); ++t) {
    const int code = (t << 4) + rit;
    const float bv = bias[code];
    v8f aHH[2], aLH[2], aHL[2];
#pragma unroll
    for (int s = 0; s < 2; ++s)
#pragma unroll
      for (int r = 0; r < 8; ++r) { aHH[s][r] = bv; aLH[s][r] = 0.f; aHL[s][r] = 0.f; }

    const v16bf* bhp = cbh + ((size_t)code << 4) + grp;   // idx = code*16 + c*2 + grp
    const v16bf* blp = cbl + ((size_t)code << 4) + grp;
#pragma unroll
    for (int c = 0; c < 8; ++c) {
      v16bf bh = bhp[c << 1];
      v16bf bl = blp[c << 1];
      aHH[0] = __builtin_amdgcn_wmma_f32_16x16x32_bf16(false, a_hi[0][c], false, bh,
                                                       (short)0, aHH[0], false, false);
      aHH[1] = __builtin_amdgcn_wmma_f32_16x16x32_bf16(false, a_hi[1][c], false, bh,
                                                       (short)0, aHH[1], false, false);
      aLH[0] = __builtin_amdgcn_wmma_f32_16x16x32_bf16(false, a_lo[0][c], false, bh,
                                                       (short)0, aLH[0], false, false);
      aLH[1] = __builtin_amdgcn_wmma_f32_16x16x32_bf16(false, a_lo[1][c], false, bh,
                                                       (short)0, aLH[1], false, false);
      aHL[0] = __builtin_amdgcn_wmma_f32_16x16x32_bf16(false, a_hi[0][c], false, bl,
                                                       (short)0, aHL[0], false, false);
      aHL[1] = __builtin_amdgcn_wmma_f32_16x16x32_bf16(false, a_hi[1][c], false, bl,
                                                       (short)0, aHL[1], false, false);
    }
#pragma unroll
    for (int s = 0; s < 2; ++s)
#pragma unroll
      for (int r = 0; r < 8; ++r) {
        float sc = aHH[s][r] + aLH[s][r] + aHL[s][r];
        bool upd = sc > best_val[s][r];
        best_val[s][r] = upd ? sc   : best_val[s][r];
        best_idx[s][r] = upd ? code : best_idx[s][r];
      }
  }

  // ---- argmax butterfly within each 16-lane half (row = r + 8*grp per strip).
  //      Tie-break: smaller index (matches argmin first-occurrence). ----
#pragma unroll
  for (int m = 1; m < 16; m <<= 1) {
#pragma unroll
    for (int s = 0; s < 2; ++s)
#pragma unroll
      for (int r = 0; r < 8; ++r) {
        float ov = __shfl_xor(best_val[s][r], m, 32);
        int   oi = __shfl_xor(best_idx[s][r], m, 32);
        bool take = (ov > best_val[s][r]) ||
                    ((ov == best_val[s][r]) && (oi < best_idx[s][r]));
        best_val[s][r] = take ? ov : best_val[s][r];
        best_idx[s][r] = take ? oi : best_idx[s][r];
      }
  }

  // ---- gather winning codes, straight-through write, loss partial ----
  float lsum = 0.f;
#pragma unroll
  for (int s = 0; s < 2; ++s) {
#pragma unroll
    for (int r = 0; r < 16; ++r) {
      int idx = __shfl(best_idx[s][r & 7], (r >> 3) << 4, 32);
      int row = row0 + (s << 4) + r;
      const float4* cv  = (const float4*)(cb  + (size_t)idx * D_DIM);
      const float4* lv  = (const float4*)(lat + (size_t)row * D_DIM);
      float4*       ovp = (float4*)      (out + (size_t)row * D_DIM);
      int j = lane << 1;                                    // 8 floats per lane
#pragma unroll
      for (int h = 0; h < 2; ++h) {
        float4 cq = cv[j + h], lq = lv[j + h], o;
        float dx = cq.x - lq.x; o.x = lq.x + dx; lsum += dx * dx;
        float dy = cq.y - lq.y; o.y = lq.y + dy; lsum += dy * dy;
        float dz = cq.z - lq.z; o.z = lq.z + dz; lsum += dz * dz;
        float dw = cq.w - lq.w; o.w = lq.w + dw; lsum += dw * dw;
        ovp[j + h] = o;
      }
    }
  }
#pragma unroll
  for (int m = 1; m < 32; m <<= 1) lsum += __shfl_xor(lsum, m, 32);
  if (lane == 0) atomicAdd(lacc, (double)lsum);
}

// ---------------------------------------------------------------------------
// Finalize: vq_loss = 1.25 * sum / N   (commitment*0.25 + embedding, same MSE)
// ---------------------------------------------------------------------------
__global__ void vq_finalize(const double* __restrict__ lacc, float* __restrict__ out) {
  out[N_FLAT] = (float)(*lacc * (1.25 / (double)N_FLAT));
}

extern "C" void kernel_launch(void* const* d_in, const int* in_sizes, int n_in,
                              void* d_out, int out_size, void* d_ws, size_t ws_size,
                              hipStream_t stream) {
  (void)in_sizes; (void)n_in; (void)out_size; (void)ws_size;
  const float* lat = (const float*)d_in[0];   // [16,4096,256] fp32
  const float* cb  = (const float*)d_in[1];   // [1024,256] fp32
  float* out = (float*)d_out;                 // 16777216 quantized + 1 loss

  char*     w    = (char*)d_ws;
  uint32_t* cbh  = (uint32_t*)w;                          // 512 KB planar hi
  uint32_t* cbl  = (uint32_t*)(w + (512 << 10));          // 512 KB planar lo
  float*    bias = (float*)(w + (1 << 20));               // 4 KB
  double*   lacc = (double*)(w + (1 << 20) + 4096);       // 8 B

  vq_pack_codebook<<<(K_CODES * D_DIM / 2) / 256, 256, 0, stream>>>(cb, cbh, cbl);
  vq_bias<<<K_CODES / 256, 256, 0, stream>>>(cb, bias, lacc);
  vq_main<<<N_ROWS / MWAVE, 32, 0, stream>>>(lat, cb, (const v16bf*)cbh,
                                             (const v16bf*)cbl, bias, out, lacc);
  vq_finalize<<<1, 1, 0, stream>>>(lacc, out);
}
